// LengthRegulator_74406013436462
// MI455X (gfx1250) — compile-verified, matrix operand order checked by reading, not run on hardware
//
#include <hip/hip_runtime.h>
#include <hip/hip_bf16.h>

// Problem constants (from reference)
#define BB      32
#define TT      512
#define CC      384
#define MAX_OUT 4096   // T * MAX_DUR

typedef __attribute__((ext_vector_type(16))) _Float16 v16h;
typedef __attribute__((ext_vector_type(8)))  float    v8f;
typedef __attribute__((ext_vector_type(4)))  float    f4;

// ---------------------------------------------------------------------------
// Kernel 1: per-batch inclusive scan of duration_target (T=512) + total.
// One block per batch, 512 threads, Hillis-Steele in LDS.
// ---------------------------------------------------------------------------
__global__ void lr_scan_kernel(const int* __restrict__ dur,
                               int* __restrict__ cum,
                               int* __restrict__ total) {
    __shared__ int s[TT];
    const int b = blockIdx.x;
    const int t = threadIdx.x;
    s[t] = dur[b * TT + t];
    __syncthreads();
    for (int off = 1; off < TT; off <<= 1) {
        int add = (t >= off) ? s[t - off] : 0;
        __syncthreads();
        s[t] += add;
        __syncthreads();
    }
    cum[b * TT + t] = s[t];
    if (t == TT - 1) total[b] = s[TT - 1];
}

// ---------------------------------------------------------------------------
// Kernel 2: src[b,pos] = searchsorted(cum[b], pos, side='right'), clipped to
// T-1; invalid (pos >= total[b]) encoded as -1. One thread per output row.
// cum is L2-resident; binary search is 9 broadcast loads per thread.
// ---------------------------------------------------------------------------
__global__ void lr_search_kernel(const int* __restrict__ cum,
                                 const int* __restrict__ total,
                                 int* __restrict__ src) {
    const int idx = blockIdx.x * blockDim.x + threadIdx.x;
    if (idx >= BB * MAX_OUT) return;
    const int b   = idx >> 12;          // / MAX_OUT
    const int pos = idx & (MAX_OUT - 1);
    if (pos >= total[b]) { src[idx] = -1; return; }
    const int* c = cum + b * TT;
    int lo = 0, hi = TT;
    while (lo < hi) {                   // upper_bound: count of c[i] <= pos
        int mid = (lo + hi) >> 1;
        if (c[mid] <= pos) lo = mid + 1; else hi = mid;
    }
    src[idx] = (lo < TT - 1) ? lo : (TT - 1);
}

// ---------------------------------------------------------------------------
// Kernel 3: the 201 MB expansion stream. One wave per output row; each lane
// moves float4, 3 iterations = 384 floats. Gathered reads hit L2 (x is 25 MB);
// the write stream uses non-temporal stores so it doesn't evict x from L2.
// 8 waves (256 threads) per block -> 16384 blocks.
// ---------------------------------------------------------------------------
__global__ void lr_expand_kernel(const float* __restrict__ x,
                                 const int* __restrict__ src,
                                 float* __restrict__ out) {
    const int row  = blockIdx.x * 8 + (threadIdx.x >> 5);   // 0 .. 131071
    const int lane = threadIdx.x & 31;
    const int s    = src[row];                              // wave-broadcast load
    const int b    = row >> 12;
    float* dst = out + (size_t)row * CC;
    if (s < 0) {
        f4 z = {0.f, 0.f, 0.f, 0.f};
#pragma unroll
        for (int it = 0; it < 3; ++it)
            __builtin_nontemporal_store(z, (f4*)(dst + it * 128 + lane * 4));
    } else {
        const float* srow = x + ((size_t)b * TT + s) * CC;
#pragma unroll
        for (int it = 0; it < 3; ++it) {
            f4 v = *(const f4*)(srow + it * 128 + lane * 4);
            __builtin_nontemporal_store(v, (f4*)(dst + it * 128 + lane * 4));
        }
    }
}

// ---------------------------------------------------------------------------
// Kernel 4: duration_pred = x @ W + b via v_wmma_f32_16x16x32_f16.
// One wave computes 16 rows. A = 16x32 f16 tile of x; B = 32x16 f16 with W's
// K-chunk replicated across all 16 columns, so every column of D is the dot
// product. 12 WMMA ops per tile (K = 384 = 12*32).
// D layout (16x16 f32): VGPR j, lane 0 -> M=j,N=0; lane 16 -> M=8+j,N=0.
// ---------------------------------------------------------------------------
__global__ void lr_pred_wmma_kernel(const float* __restrict__ x,
                                    const float* __restrict__ W,
                                    const float* __restrict__ bias,
                                    float* __restrict__ out) {
    const int lane     = threadIdx.x & 31;
    const int wave     = threadIdx.x >> 5;
    const int tile     = blockIdx.x * 4 + wave;   // 16 rows per tile
    const int row_base = tile * 16;               // < B*T = 16384
    const int m        = lane & 15;
    const int off      = (lane < 16) ? 0 : 8;     // K sub-block per lane group

    const float* xrow = x + (size_t)(row_base + m) * CC;

    v8f c = {};
#pragma unroll
    for (int kb = 0; kb < CC; kb += 32) {
        v16h a, bm;
#pragma unroll
        for (int e = 0; e < 8; ++e) {
            // A-matrix f16 16x32 layout: halves 0..7 -> K=off+e,
            //                            halves 8..15 -> K=off+16+e
            a[e]      = (_Float16)xrow[kb + off + e];
            a[e + 8]  = (_Float16)xrow[kb + off + 16 + e];
            // B-matrix: same K mapping, W replicated over all N columns
            bm[e]     = (_Float16)W[kb + off + e];
            bm[e + 8] = (_Float16)W[kb + off + 16 + e];
        }
        c = __builtin_amdgcn_wmma_f32_16x16x32_f16(
                /*neg_a=*/false, a, /*neg_b=*/false, bm,
                /*c_mod=*/(short)0, c, /*reuse_a=*/false, /*reuse_b=*/false);
    }

    const float bv = bias[0];
    if (lane == 0) {
#pragma unroll
        for (int j = 0; j < 8; ++j) out[row_base + j] = c[j] + bv;
    } else if (lane == 16) {
#pragma unroll
        for (int j = 0; j < 8; ++j) out[row_base + 8 + j] = c[j] + bv;
    }
}

// ---------------------------------------------------------------------------
// Launcher. Inputs: x (f32, B*T*C), W (f32, C), b (f32, 1), duration (i32, B*T).
// Output: expanded (B*MAX_OUT*C f32) ++ duration_pred (B*T f32).
// Workspace: cum (B*T i32) | total (B i32, padded) | src (B*MAX_OUT i32).
// ---------------------------------------------------------------------------
extern "C" void kernel_launch(void* const* d_in, const int* in_sizes, int n_in,
                              void* d_out, int out_size, void* d_ws, size_t ws_size,
                              hipStream_t stream) {
    const float* x    = (const float*)d_in[0];
    const float* W    = (const float*)d_in[1];
    const float* bptr = (const float*)d_in[2];
    const int*   dur  = (const int*)d_in[3];

    float* out_expanded = (float*)d_out;
    float* out_pred     = out_expanded + (size_t)BB * MAX_OUT * CC;

    int* cum   = (int*)d_ws;              // B*T ints      (64 KB)
    int* total = cum + BB * TT;           // B ints
    int* src   = total + 64;              // B*MAX_OUT ints (512 KB), padded start

    // 1) per-batch cumsum + totals
    lr_scan_kernel<<<BB, TT, 0, stream>>>(dur, cum, total);

    // 2) searchsorted for every output position
    lr_search_kernel<<<(BB * MAX_OUT + 255) / 256, 256, 0, stream>>>(cum, total, src);

    // 3) bandwidth-bound expansion (201 MB NT store stream)
    lr_expand_kernel<<<(BB * MAX_OUT) / 8, 256, 0, stream>>>(x, src, out_expanded);

    // 4) duration_pred GEMV on the WMMA path: 16384 rows / 16 per wave / 4 waves
    lr_pred_wmma_kernel<<<(BB * TT) / (16 * 4), 128, 0, stream>>>(x, W, bptr, out_pred);
}
